// RwkvTimeMix_16183436772029
// MI455X (gfx1250) — matrix operand order repaired and checked by reading, not compiled, not run on hardware
//
#include <hip/hip_runtime.h>

typedef _Float16 f16;
typedef __attribute__((ext_vector_type(16))) _Float16 v16h;
typedef __attribute__((ext_vector_type(8)))  _Float16 v8h;
typedef __attribute__((ext_vector_type(4)))  _Float16 v4h;
typedef __attribute__((ext_vector_type(8)))  float    v8f;
typedef __attribute__((ext_vector_type(4)))  float    v4f;

#define HDIM 1024
#define BDIM 16384
#define BH   ((size_t)BDIM * (size_t)HDIM)   // 16,777,216 elements
#define HH   ((size_t)HDIM * (size_t)HDIM)   // 1,048,576 elements

// LDS pitch for a 64-half weight row; 72 halfs (144 B) de-staggers the 64
// 4-byte banks (stride 36 banks, gcd(36,64)=4) while staying 16B-aligned.
#define LPITCH 72

// ---------------------------------------------------------------------------
// WMMA helper (CDNA5 v_wmma_f32_16x16x32_f16, wave32)
// ---------------------------------------------------------------------------
__device__ __forceinline__ v8f wmma_f32_16x16x32(v16h a, v16h b, v8f c) {
    // 8 args: (neg_a, A, neg_b, B, c_mod, C, reuse_a, reuse_b)
    return __builtin_amdgcn_wmma_f32_16x16x32_f16(false, a, false, b,
                                                  (short)0, c, false, false);
}

// A 16x32 f16 tile in this code's k order (same as B): lane l<16 -> M=l, K in {0..15};
// lane l>=16 -> M=l-16, K in {16..31}.  Two 16B loads, K-contiguous rows.
__device__ __forceinline__ v16h load_a_tile(const f16* __restrict__ row,
                                            int k0, int hi) {
    const v8h lo = *(const v8h*)(row + k0 + hi * 16);
    const v8h hh = *(const v8h*)(row + k0 + hi * 16 + 8);
    v16h a;
#pragma unroll
    for (int i = 0; i < 8; ++i) { a[i] = lo[i]; a[i + 8] = hh[i]; }
    return a;
}

// B 32x16 f16 tile in this code's k order (same as A): lane n holds column n; v16h elements = 16
// K-consecutive halfs starting at k0 + 16*(lane>=16).  Read from LDS as two
// aligned 16B v8h loads (LPITCH*2 = 144 B is 16B- but not 32B-aligned).
__device__ __forceinline__ v16h load_b_lds(const f16* lds, int col, int k) {
    const f16* p = lds + col * LPITCH + k;
    const v8h lo = *(const v8h*)(p);
    const v8h hh = *(const v8h*)(p + 8);
    v16h b;
#pragma unroll
    for (int i = 0; i < 8; ++i) { b[i] = lo[i]; b[i + 8] = hh[i]; }
    return b;
}

// ---------------------------------------------------------------------------
// Kernel 1: token-shift mix + f32->f16 cast; also emits output #2 (x copy)
// ---------------------------------------------------------------------------
__global__ __launch_bounds__(256) void mix_cast_kernel(
    const float* __restrict__ x, const float* __restrict__ alpha,
    const float* __restrict__ tmk, const float* __restrict__ tmv,
    const float* __restrict__ tmr,
    f16* __restrict__ xk16, f16* __restrict__ xv16, f16* __restrict__ xr16,
    float* __restrict__ x_out)
{
    size_t base = ((size_t)blockIdx.x * blockDim.x + threadIdx.x) * 4;
    if (base >= BH) return;
    const int col = (int)(base % HDIM);

    v4f xv4 = *(const v4f*)(x + base);
    v4f av4 = *(const v4f*)(alpha + base);
    v4f mk  = *(const v4f*)(tmk + col);
    v4f mv  = *(const v4f*)(tmv + col);
    v4f mr  = *(const v4f*)(tmr + col);

    v4h ok, ov, orr;
#pragma unroll
    for (int i = 0; i < 4; ++i) {
        ok[i]  = (f16)(xv4[i] * mk[i] + av4[i] * (1.0f - mk[i]));
        ov[i]  = (f16)(xv4[i] * mv[i] + av4[i] * (1.0f - mv[i]));
        orr[i] = (f16)(xv4[i] * mr[i] + av4[i] * (1.0f - mr[i]));
    }
    *(v4h*)(xk16 + base) = ok;
    *(v4h*)(xv16 + base) = ov;
    *(v4h*)(xr16 + base) = orr;
    *(v4f*)(x_out + base) = xv4;   // reference output #2 is x unchanged
}

// ---------------------------------------------------------------------------
// Kernel 2: weights f32 -> f16 (8 MB total; stays resident in 192 MB L2)
// ---------------------------------------------------------------------------
__global__ __launch_bounds__(256) void wconv_kernel(
    const float* __restrict__ wk, const float* __restrict__ wv,
    const float* __restrict__ wr, const float* __restrict__ wo,
    f16* __restrict__ wk16, f16* __restrict__ wv16,
    f16* __restrict__ wr16, f16* __restrict__ wo16)
{
    size_t base = ((size_t)blockIdx.x * blockDim.x + threadIdx.x) * 4;
    if (base >= 4 * HH) return;
    const int which = (int)(base / HH);
    const size_t off = base % HH;
    const float* src = (which == 0) ? wk : (which == 1) ? wv
                                         : (which == 2) ? wr : wo;
    f16* dst = (which == 0) ? wk16 : (which == 1) ? wv16
                                   : (which == 2) ? wr16 : wo16;
    v4f s = *(const v4f*)(src + off);
    v4h d;
#pragma unroll
    for (int i = 0; i < 4; ++i) d[i] = (f16)s[i];
    *(v4h*)(dst + off) = d;
}

// ---------------------------------------------------------------------------
// Kernel 3: fused triple-GEMM (k, v, r) + WKV epilogue.
// Block tile: 128(M) x 64(N); the 8 waves share one 64-column group so the
// weight K-chunk (3 x 64 x 64 halfs) is staged into LDS once per block and
// consumed 8x via ds loads.  Wave tile: 16x64, 12 accumulators, 12 WMMA/K-step.
// ---------------------------------------------------------------------------
__global__ __launch_bounds__(256) void fused_kvr_wkv_kernel(
    const f16* __restrict__ xk16, const f16* __restrict__ xv16,
    const f16* __restrict__ xr16,
    const f16* __restrict__ wk16, const f16* __restrict__ wv16,
    const f16* __restrict__ wr16,
    const float* __restrict__ aa, const float* __restrict__ bb,
    const float* __restrict__ pp,
    const float* __restrict__ time_first, const float* __restrict__ time_decay,
    float* __restrict__ next_aa, float* __restrict__ next_bb,
    float* __restrict__ qq2_out, f16* __restrict__ rwkv16)
{
    __shared__ f16 lbs[3][64 * LPITCH];          // ~27 KB of the 320 KB LDS

    const int wave = threadIdx.x >> 5;           // 0..7 -> M sub-tile
    const int lane = threadIdx.x & 31;
    const int l16  = lane & 15;
    const int hi   = lane >> 4;

    const int mBlock = blockIdx.x >> 4;          // 128 row-blocks of 128
    const int nGroup = blockIdx.x & 15;          // 16 column groups of 64
    const int m0 = mBlock * 128 + wave * 16;
    const int n0 = nGroup * 64;

    v8f accK[4], accV[4], accR[4];
#pragma unroll
    for (int t = 0; t < 4; ++t) {
        accK[t] = (v8f)(0.0f); accV[t] = (v8f)(0.0f); accR[t] = (v8f)(0.0f);
    }

    const f16* aK = xk16 + (size_t)(m0 + l16) * HDIM;
    const f16* aV = xv16 + (size_t)(m0 + l16) * HDIM;
    const f16* aR = xr16 + (size_t)(m0 + l16) * HDIM;

    for (int kc = 0; kc < HDIM; kc += 64) {
        __syncthreads();                          // chunk consumed by all waves
        // Cooperative stage: 3 mats x 64 cols x 8 chunks of 8 halfs = 1536
        // 16B transfers; 256 threads x 6.
#pragma unroll
        for (int i = 0; i < 6; ++i) {
            const int cid    = threadIdx.x + i * 256;
            const int mat    = cid >> 9;          // /512
            const int rem    = cid & 511;
            const int colIdx = rem >> 3;
            const int ko     = (rem & 7) * 8;
            const f16* src = (mat == 0 ? wk16 : mat == 1 ? wv16 : wr16)
                             + (size_t)(n0 + colIdx) * HDIM + kc + ko;
            *(v8h*)&lbs[mat][colIdx * LPITCH + ko] = *(const v8h*)src;
        }
        __syncthreads();                          // staging visible

        if (kc + 128 < HDIM) {                    // gfx1250 global_prefetch_b8
            __builtin_prefetch(aK + kc + 128, 0, 1);
            __builtin_prefetch(aV + kc + 128, 0, 1);
            __builtin_prefetch(aR + kc + 128, 0, 1);
        }

#pragma unroll
        for (int kr = 0; kr < 64; kr += 32) {
            const v16h avK = load_a_tile(aK, kc + kr, hi);
            const v16h avV = load_a_tile(aV, kc + kr, hi);
            const v16h avR = load_a_tile(aR, kc + kr, hi);
            const int kb = kr + hi * 16;
#pragma unroll
            for (int t = 0; t < 4; ++t) {
                const int col = t * 16 + l16;
                accK[t] = wmma_f32_16x16x32(avK, load_b_lds(lbs[0], col, kb), accK[t]);
                accV[t] = wmma_f32_16x16x32(avV, load_b_lds(lbs[1], col, kb), accV[t]);
                accR[t] = wmma_f32_16x16x32(avR, load_b_lds(lbs[2], col, kb), accR[t]);
            }
        }
    }

    // Epilogue: C/D layout -> element r of v8f is row (m0 + r + 8*hi), col n0+16t+l16
#pragma unroll
    for (int t = 0; t < 4; ++t) {
        const int n = n0 + t * 16 + l16;
        const float tf = time_first[n];
        const float td = time_decay[n];
#pragma unroll
        for (int r = 0; r < 8; ++r) {
            const int m = m0 + r + 8 * hi;
            const size_t idx = (size_t)m * HDIM + n;
            const float kk = accK[t][r];
            const float vv = accV[t][r];
            const float rr = accR[t][r];
            const float paa = aa[idx], pbb = bb[idx], ppp = pp[idx];
            // wkv with log-sum-exp stabilization
            const float ww  = tf + kk;
            const float qq  = fmaxf(ppp, ww);
            const float e1  = __expf(ppp - qq);
            const float e2  = __expf(ww - qq);
            const float wkv = (e1 * paa + e2 * vv) / (e1 * pbb + e2);
            // state update with decay
            const float ww2 = ppp + td;
            const float q2  = fmaxf(ww2, kk);
            const float e1b = __expf(ww2 - q2);
            const float e2b = __expf(kk - q2);
            next_aa[idx] = e1b * paa + e2b * vv;
            next_bb[idx] = e1b * pbb + e2b;
            qq2_out[idx] = q2;
            const float rs = 1.0f / (1.0f + __expf(-rr));
            rwkv16[idx] = (f16)(rs * wkv);
        }
    }
}

// ---------------------------------------------------------------------------
// Kernel 4: out = (r*wkv) @ Wo^T  (single GEMM, same LDS-staged tiling)
// ---------------------------------------------------------------------------
__global__ __launch_bounds__(256) void out_gemm_kernel(
    const f16* __restrict__ rwkv16, const f16* __restrict__ wo16,
    float* __restrict__ out)
{
    __shared__ f16 lbo[64 * LPITCH];             // ~9 KB

    const int wave = threadIdx.x >> 5;
    const int lane = threadIdx.x & 31;
    const int l16  = lane & 15;
    const int hi   = lane >> 4;

    const int mBlock = blockIdx.x >> 4;
    const int nGroup = blockIdx.x & 15;
    const int m0 = mBlock * 128 + wave * 16;
    const int n0 = nGroup * 64;

    v8f acc[4];
#pragma unroll
    for (int t = 0; t < 4; ++t) acc[t] = (v8f)(0.0f);

    const f16* aRow = rwkv16 + (size_t)(m0 + l16) * HDIM;

    for (int kc = 0; kc < HDIM; kc += 64) {
        __syncthreads();
        // 64 cols x 8 chunks = 512 16B transfers; 256 threads x 2.
#pragma unroll
        for (int i = 0; i < 2; ++i) {
            const int cid    = threadIdx.x + i * 256;
            const int colIdx = cid >> 3;
            const int ko     = (cid & 7) * 8;
            const f16* src = wo16 + (size_t)(n0 + colIdx) * HDIM + kc + ko;
            *(v8h*)&lbo[colIdx * LPITCH + ko] = *(const v8h*)src;
        }
        __syncthreads();

        if (kc + 128 < HDIM) __builtin_prefetch(aRow + kc + 128, 0, 1);

#pragma unroll
        for (int kr = 0; kr < 64; kr += 32) {
            const v16h av = load_a_tile(aRow, kc + kr, hi);
            const int kb = kr + hi * 16;
#pragma unroll
            for (int t = 0; t < 4; ++t) {
                acc[t] = wmma_f32_16x16x32(av, load_b_lds(lbo, t * 16 + l16, kb), acc[t]);
            }
        }
    }

#pragma unroll
    for (int t = 0; t < 4; ++t) {
        const int n = n0 + t * 16 + l16;
#pragma unroll
        for (int r = 0; r < 8; ++r) {
            const int m = m0 + r + 8 * hi;
            out[(size_t)m * HDIM + n] = acc[t][r];
        }
    }
}

// ---------------------------------------------------------------------------
// Launch: mix -> wconv -> fused(k,v,r,wkv) -> out GEMM, all on `stream`.
// Workspace layout (136 MB): xk16 | xv16 | xr16 | wk16 | wv16 | wr16 | wo16 | rwkv16
// ---------------------------------------------------------------------------
extern "C" void kernel_launch(void* const* d_in, const int* in_sizes, int n_in,
                              void* d_out, int out_size, void* d_ws, size_t ws_size,
                              hipStream_t stream) {
    const float* x     = (const float*)d_in[0];
    const float* alpha = (const float*)d_in[1];
    const float* aa    = (const float*)d_in[2];
    const float* bb    = (const float*)d_in[3];
    const float* pp    = (const float*)d_in[4];
    const float* tdec  = (const float*)d_in[5];
    const float* tfirst= (const float*)d_in[6];
    const float* tmk   = (const float*)d_in[7];
    const float* tmv   = (const float*)d_in[8];
    const float* tmr   = (const float*)d_in[9];
    const float* Wk    = (const float*)d_in[10];
    const float* Wv    = (const float*)d_in[11];
    const float* Wr    = (const float*)d_in[12];
    const float* Wo    = (const float*)d_in[13];

    float* out = (float*)d_out;          // [out | x | next_aa | next_bb | qq2]
    float* x_out    = out + BH;
    float* next_aa  = out + 2 * BH;
    float* next_bb  = out + 3 * BH;
    float* qq2_out  = out + 4 * BH;

    char* ws = (char*)d_ws;
    f16* xk16   = (f16*)(ws);
    f16* xv16   = (f16*)(ws + 2 * BH);
    f16* xr16   = (f16*)(ws + 4 * BH);
    f16* wk16   = (f16*)(ws + 6 * BH);
    f16* wv16   = (f16*)(ws + 6 * BH + 2 * HH);
    f16* wr16   = (f16*)(ws + 6 * BH + 4 * HH);
    f16* wo16   = (f16*)(ws + 6 * BH + 6 * HH);
    f16* rwkv16 = (f16*)(ws + 6 * BH + 8 * HH);

    // 1) mix + cast (BH/4 lanes of float4)
    mix_cast_kernel<<<(unsigned)(BH / 4 / 256), 256, 0, stream>>>(
        x, alpha, tmk, tmv, tmr, xk16, xv16, xr16, x_out);

    // 2) weight conversion (4*HH/4 lanes)
    wconv_kernel<<<(unsigned)(4 * HH / 4 / 256), 256, 0, stream>>>(
        Wk, Wv, Wr, Wo, wk16, wv16, wr16, wo16);

    // 3) fused k/v/r GEMMs + WKV: 128 row-blocks x 16 col-groups = 2048 blocks
    fused_kvr_wkv_kernel<<<2048, 256, 0, stream>>>(
        xk16, xv16, xr16, wk16, wv16, wr16,
        aa, bb, pp, tfirst, tdec,
        next_aa, next_bb, qq2_out, rwkv16);

    // 4) output GEMM
    out_gemm_kernel<<<2048, 256, 0, stream>>>(rwkv16, wo16, out);
}
